// QuantBertSelfAttention_39256001085976
// MI455X (gfx1250) — matrix operand order repaired and checked by reading, not compiled, not run on hardware
//
#include <hip/hip_runtime.h>
#include <hip/hip_bf16.h>

// MI455X / gfx1250, wave32. WMMA f16 path: all mxfp-quantized operands
// (E2M1 x power-of-two scale) are exactly representable in f16.

typedef __attribute__((ext_vector_type(16))) _Float16 v16h;
typedef __attribute__((ext_vector_type(8)))  float    v8f;
typedef int v4i __attribute__((vector_size(16)));   // b128 payload type

#define WMMA_F16(a, b, c) \
  __builtin_amdgcn_wmma_f32_16x16x32_f16(false, (a), false, (b), (short)0, (c), false, false)

// ---- gfx1250 async global->LDS staging (ASYNCcnt path), guarded -------------
#if defined(__has_builtin)
#  if __has_builtin(__builtin_amdgcn_global_load_async_to_lds_b128)
#    define ASYNC_LDS 1
#  endif
#endif
#ifndef ASYNC_LDS
#  define ASYNC_LDS 0
#endif

__device__ __forceinline__ void lds_copy_b128(void* lds_dst, const void* gsrc) {
#if ASYNC_LDS
  __builtin_amdgcn_global_load_async_to_lds_b128(
      (__attribute__((address_space(1))) v4i*)(uintptr_t)gsrc,
      (__attribute__((address_space(3))) v4i*)lds_dst, 0, 0);
#else
  *(v4i*)lds_dst = *(const v4i*)gsrc;
#endif
}

__device__ __forceinline__ void lds_async_wait0() {
#if ASYNC_LDS
#  if defined(__has_builtin) && __has_builtin(__builtin_amdgcn_s_wait_asynccnt)
  __builtin_amdgcn_s_wait_asynccnt(0);
#  else
  asm volatile("s_wait_asynccnt 0x0" ::: "memory");
#  endif
#endif
}

// ---------------- MXFP (E2M1, GS=32, POT scale) quant helpers ----------------
__device__ __forceinline__ float mxfp_scale(float amax) {
  if (amax == 0.0f) amax = 1.0f;
  float pot = floorf(log2f(amax) - 2.0f);     // EXP_W=2 -> 2^(EXP_W-1)=2
  return fminf(exp2f(pot), 255.0f);
}

__device__ __forceinline__ float mxfp_val(float xd) {
  if (xd == 0.0f) return 0.0f;
  float sgn = (xd < 0.0f) ? -1.0f : 1.0f;
  float xa  = fabsf(xd);
  float e   = floorf(log2f(xa));
  float p2e = exp2f(e);
  float man = xa / p2e;
  float xr  = rintf(man * 2.0f) * 0.5f * p2e; // MAN_W=1, round-half-even
  float xc  = fminf(fmaxf(xr, 0.5f), 6.0f);   // min_repr=0.5, max_repr=6
  if (xa <= 0.25f) xc = 0.0f;                 // flush below min_repr/2
  return sgn * xc;
}

__device__ __forceinline__ float mxfp_fq(float x, float sc) {
  return mxfp_val(x / sc) * sc;
}

// ---------------- Kernel 0: one-shot f32 -> f16 downconvert ------------------
__global__ void __launch_bounds__(256)
f32_to_f16(const float* __restrict__ in, _Float16* __restrict__ outp, int n) {
  int i = (blockIdx.x * 256 + threadIdx.x) * 4;
  if (i + 3 < n) {
#pragma unroll
    for (int j = 0; j < 4; ++j) outp[i + j] = (_Float16)in[i + j];
  }
}

// ---------------- Kernel 1: QKV projection GEMM (f16 in, f32 out) ------------
// C[m,n] = sum_k X[m,k] * W[n,k] + b[n].  M=4096, N=K=1024.
// Block 256 thr (8 waves = 4Mx2N), tile 128x64, k-step 32.
// LDS staged via async b128 copies; row stride 40 halves = 80 B (16B aligned,
// 20-dword bank stride -> conflict-free 16-lane fragment gathers).
#define LDA 40
#define LDB 40

__global__ void __launch_bounds__(256)
qkv_gemm(const _Float16* __restrict__ Xh, const _Float16* __restrict__ Wh,
         const float* __restrict__ bq, const float* __restrict__ bk,
         const float* __restrict__ bv,
         float* __restrict__ Qf, float* __restrict__ Kf, float* __restrict__ Vf) {
  __shared__ _Float16 As[128 * LDA];
  __shared__ _Float16 Bs[64 * LDB];

  const _Float16* W = Wh + (size_t)blockIdx.z * 1024 * 1024;
  const float* bias = (blockIdx.z == 0) ? bq : (blockIdx.z == 1) ? bk : bv;
  float* out        = (blockIdx.z == 0) ? Qf : (blockIdx.z == 1) ? Kf : Vf;

  const int tid   = threadIdx.x;
  const int wv    = tid >> 5;
  const int lane  = tid & 31;
  const int kh    = lane >> 4;     // lane half
  const int lm    = lane & 15;
  const int waveM = wv & 3;        // 4 waves along M
  const int waveN = wv >> 2;       // 2 waves along N
  const int m0    = blockIdx.y * 128;
  const int n0    = blockIdx.x * 64;

  v8f acc[2][2] = {{{}, {}}, {{}, {}}};

  for (int k0 = 0; k0 < 1024; k0 += 32) {
    __syncthreads();
    // Stage A: 128x32 halves = 512 x 16B chunks (2 per thread), async -> LDS
#pragma unroll
    for (int cc = 0; cc < 2; ++cc) {
      int chunk = tid + cc * 256;
      int r  = chunk >> 2;
      int cs = (chunk & 3) * 8;
      const _Float16* src = Xh + (size_t)(m0 + r) * 1024 + k0 + cs;
      lds_copy_b128(&As[r * LDA + cs], src);
      if (k0 + 32 < 1024) __builtin_prefetch(src + 32, 0, 0);
    }
    // Stage B: 64x32 halves = 256 x 16B chunks (1 per thread), async -> LDS
    {
      int r  = tid >> 2;
      int cs = (tid & 3) * 8;
      lds_copy_b128(&Bs[r * LDB + cs], W + (size_t)(n0 + r) * 1024 + k0 + cs);
    }
    lds_async_wait0();   // this wave's async LDS writes complete
    __syncthreads();     // all waves' tiles visible

    v16h af[2], bf[2];
#pragma unroll
    for (int i = 0; i < 2; ++i) {
      int mrow = waveM * 32 + i * 16 + lm;
#pragma unroll
      for (int r = 0; r < 8; ++r) {
        int kb = 2 * r + ((r >= 4) ? 8 : 0) + 8 * kh;   // A 16x32 f16 layout
        af[i][2 * r]     = As[mrow * LDA + kb];
        af[i][2 * r + 1] = As[mrow * LDA + kb + 1];
      }
    }
#pragma unroll
    for (int j = 0; j < 2; ++j) {
      int ncol = waveN * 32 + j * 16 + lm;
#pragma unroll
      for (int r = 0; r < 8; ++r) {
        int kb = 2 * r + ((r >= 4) ? 8 : 0) + 8 * kh;   // B 32x16 f16 layout
        bf[j][2 * r]     = Bs[ncol * LDB + kb];
        bf[j][2 * r + 1] = Bs[ncol * LDB + kb + 1];
      }
    }
#pragma unroll
    for (int i = 0; i < 2; ++i)
#pragma unroll
      for (int j = 0; j < 2; ++j)
        acc[i][j] = WMMA_F16(af[i], bf[j], acc[i][j]);
  }

  // Epilogue: bias + store f32
#pragma unroll
  for (int i = 0; i < 2; ++i) {
#pragma unroll
    for (int j = 0; j < 2; ++j) {
      int n = n0 + waveN * 32 + j * 16 + lm;
      float bvv = bias[n];
#pragma unroll
      for (int r = 0; r < 8; ++r) {
        int m = m0 + waveM * 32 + i * 16 + r + 8 * kh;   // C layout
        out[(size_t)m * 1024 + n] = acc[i][j][r] + bvv;
      }
    }
  }
}

// ---------------- Kernel 2: per-32-group MXFP fake-quant, f32 -> f16 ---------
__global__ void __launch_bounds__(256)
quant_groups(const float* __restrict__ in, _Float16* __restrict__ outp, int ngroups) {
  int g = blockIdx.x * blockDim.x + threadIdx.x;
  if (g >= ngroups) return;
  const float* p = in + (size_t)g * 32;
  float v[32];
  float amax = 0.0f;
#pragma unroll
  for (int i = 0; i < 32; ++i) { v[i] = p[i]; amax = fmaxf(amax, fabsf(v[i])); }
  float sc = mxfp_scale(amax);
  _Float16* o = outp + (size_t)g * 32;
#pragma unroll
  for (int i = 0; i < 32; ++i) o[i] = (_Float16)mxfp_fq(v[i], sc);
}

// ---------------- Kernel 3: fused attention per (b, h, 16 query rows) --------
// scores(WMMA) -> quant -> /8 + mask -> softmax -> quant -> PV(WMMA) -> quant
#define SLD 516   // padded f32 stride: (516 mod 64)=4 -> conflict-free gathers

__global__ void __launch_bounds__(256)
attn_fused(const _Float16* __restrict__ Qh, const _Float16* __restrict__ Kh,
           const _Float16* __restrict__ Vh, const float* __restrict__ mask,
           float* __restrict__ out) {
  __shared__ float sS[16 * SLD];        // 33 KB score/prob panel
  __shared__ float sRed[2][16 * 16];    // row max / row sum partials
  __shared__ float sCtx[2][16 * 64];    // split-K PV partials

  const int bh  = blockIdx.y;
  const int b   = bh >> 4;
  const int h   = bh & 15;
  const int q0  = blockIdx.x * 16;
  const int tid = threadIdx.x;
  const int wv  = tid >> 5;
  const int lane = tid & 31;
  const int kh  = lane >> 4;
  const int lm  = lane & 15;
  const size_t baseBH = (size_t)b * 512 * 1024 + (size_t)h * 64; // + s*1024 + d

  // ---- Phase 1: scores[16 x 512] = q @ k^T (wave wv covers cols wv*64..+63)
  {
    v8f c[4] = {{}, {}, {}, {}};
    for (int kk = 0; kk < 64; kk += 32) {
      v16h a;
      const _Float16* qp = Qh + baseBH + (size_t)(q0 + lm) * 1024 + kk;
#pragma unroll
      for (int r = 0; r < 8; ++r) {
        int kb = 2 * r + ((r >= 4) ? 8 : 0) + 8 * kh;
        a[2 * r]     = qp[kb];
        a[2 * r + 1] = qp[kb + 1];
      }
#pragma unroll
      for (int j = 0; j < 4; ++j) {
        int ncol = wv * 64 + j * 16 + lm;            // key row index
        const _Float16* kp = Kh + baseBH + (size_t)ncol * 1024 + kk;
        v16h bfr;
#pragma unroll
        for (int r = 0; r < 8; ++r) {
          int kd = 2 * r + ((r >= 4) ? 8 : 0) + 8 * kh;
          bfr[2 * r]     = kp[kd];
          bfr[2 * r + 1] = kp[kd + 1];
        }
        c[j] = WMMA_F16(a, bfr, c[j]);
      }
    }
#pragma unroll
    for (int j = 0; j < 4; ++j)
#pragma unroll
      for (int r = 0; r < 8; ++r)
        sS[(r + 8 * kh) * SLD + wv * 64 + j * 16 + lm] = c[j][r];
  }
  __syncthreads();

  // ---- Phase 2: quant scores -> scale+mask -> softmax -> quant probs
  {
    const int row = tid >> 4;
    const int g   = tid & 15;
    const int cb  = g * 32;
    float* rowp = sS + row * SLD + cb;
    const float* mp = mask + (size_t)b * 512 + cb;
    float vals[32];
    float amax = 0.0f;
#pragma unroll
    for (int i = 0; i < 32; ++i) { vals[i] = rowp[i]; amax = fmaxf(amax, fabsf(vals[i])); }
    float sc = mxfp_scale(amax);
    float lmax = -3.0e38f;
#pragma unroll
    for (int i = 0; i < 32; ++i) {
      float s = mxfp_fq(vals[i], sc) * 0.125f + mp[i]; // 1/sqrt(64)=0.125
      vals[i] = s;
      lmax = fmaxf(lmax, s);
    }
    sRed[0][row * 16 + g] = lmax;
    __syncthreads();
    float rmax = -3.0e38f;
#pragma unroll
    for (int t = 0; t < 16; ++t) rmax = fmaxf(rmax, sRed[0][row * 16 + t]);
    float lsum = 0.0f;
#pragma unroll
    for (int i = 0; i < 32; ++i) { vals[i] = __expf(vals[i] - rmax); lsum += vals[i]; }
    sRed[1][row * 16 + g] = lsum;
    __syncthreads();
    float rsum = 0.0f;
#pragma unroll
    for (int t = 0; t < 16; ++t) rsum += sRed[1][row * 16 + t];
    float inv = 1.0f / rsum;
    float pmax = 0.0f;
#pragma unroll
    for (int i = 0; i < 32; ++i) { vals[i] *= inv; pmax = fmaxf(pmax, vals[i]); }
    float sc2 = mxfp_scale(pmax);
#pragma unroll
    for (int i = 0; i < 32; ++i) rowp[i] = mxfp_fq(vals[i], sc2); // in-place f32
  }
  __syncthreads();

  // ---- Phase 3: ctx[16 x 64] = probs @ V ; split-K: 4 N-tiles x 2 K-halves
  {
    const int ntile = wv & 3;
    const int khalf = wv >> 2;
    v8f c = {};
    for (int ks = 0; ks < 8; ++ks) {
      int k0 = (khalf * 8 + ks) * 32;
      v16h a;
#pragma unroll
      for (int r = 0; r < 8; ++r) {
        int kb = k0 + 2 * r + ((r >= 4) ? 8 : 0) + 8 * kh;
        a[2 * r]     = (_Float16)sS[lm * SLD + kb];
        a[2 * r + 1] = (_Float16)sS[lm * SLD + kb + 1];
      }
      v16h bfr;
      int n = ntile * 16 + lm;                       // dh index
#pragma unroll
      for (int r = 0; r < 8; ++r) {
        int sidx = k0 + 2 * r + ((r >= 4) ? 8 : 0) + 8 * kh; // key/value row
        const _Float16* vp = Vh + baseBH + (size_t)sidx * 1024 + n;
        bfr[2 * r]     = vp[0];
        bfr[2 * r + 1] = vp[1024];                   // next row, same dh
      }
      c = WMMA_F16(a, bfr, c);
    }
#pragma unroll
    for (int r = 0; r < 8; ++r)
      sCtx[khalf][(r + 8 * kh) * 64 + ntile * 16 + lm] = c[r];
  }
  __syncthreads();

  // ---- Phase 4: combine split-K, MXFP-quant ctx (2 groups/row), store f32
  if (tid < 32) {
    int r4  = tid >> 1;
    int cb4 = (tid & 1) * 32;
    float v4[32];
    float am = 0.0f;
#pragma unroll
    for (int i = 0; i < 32; ++i) {
      float x = sCtx[0][r4 * 64 + cb4 + i] + sCtx[1][r4 * 64 + cb4 + i];
      v4[i] = x;
      am = fmaxf(am, fabsf(x));
    }
    float s4 = mxfp_scale(am);
    float* op = out + (size_t)(b * 512 + q0 + r4) * 1024 + (size_t)h * 64 + cb4;
#pragma unroll
    for (int i = 0; i < 32; ++i) op[i] = mxfp_fq(v4[i], s4);
  }
}

// ---------------- Host launcher ----------------------------------------------
extern "C" void kernel_launch(void* const* d_in, const int* in_sizes, int n_in,
                              void* d_out, int out_size, void* d_ws, size_t ws_size,
                              hipStream_t stream) {
  const float* X    = (const float*)d_in[0];
  const float* mask = (const float*)d_in[1];
  const float* Wq   = (const float*)d_in[2];
  const float* bq   = (const float*)d_in[3];
  const float* Wk   = (const float*)d_in[4];
  const float* bk   = (const float*)d_in[5];
  const float* Wv   = (const float*)d_in[6];
  const float* bv   = (const float*)d_in[7];
  float* out = (float*)d_out;

  const size_t NE = (size_t)8 * 512 * 1024;   // 4 Mi elements per activation
  const size_t NW = (size_t)1024 * 1024;      // 1 Mi elements per weight
  char* ws = (char*)d_ws;
  float* Qf = (float*)ws;                      // 3 contiguous f32 tensors
  float* Kf = Qf + NE;
  float* Vf = Kf + NE;
  _Float16* Qh = (_Float16*)(ws + 3 * NE * sizeof(float));  // 3 contiguous f16
  _Float16* Kh = Qh + NE;
  _Float16* Vh = Kh + NE;
  _Float16* Xh = Vh + NE;                      // f16 copy of hidden_states
  _Float16* Wh = Xh + NE;                      // f16 Wq|Wk|Wv contiguous

  // 0) One-shot operand downconvert (halves GEMM operand traffic)
  f32_to_f16<<<(int)(NE / 4 / 256), 256, 0, stream>>>(X, Xh, (int)NE);
  f32_to_f16<<<(int)(NW / 4 / 256), 256, 0, stream>>>(Wq, Wh,          (int)NW);
  f32_to_f16<<<(int)(NW / 4 / 256), 256, 0, stream>>>(Wk, Wh + NW,     (int)NW);
  f32_to_f16<<<(int)(NW / 4 / 256), 256, 0, stream>>>(Wv, Wh + 2 * NW, (int)NW);

  // 1) QKV projections: grid.z selects Q/K/V
  dim3 g1(1024 / 64, 4096 / 128, 3);
  qkv_gemm<<<g1, 256, 0, stream>>>(Xh, Wh, bq, bk, bv, Qf, Kf, Vf);

  // 2) MXFP fake-quant q,k,v (contiguous groups of 32 along Dh)
  int ngroups = (int)(3 * NE / 32);            // 393216
  quant_groups<<<(ngroups + 255) / 256, 256, 0, stream>>>(Qf, Qh, ngroups);

  // 3) Fused attention: one WG per (16 query rows, b*h)
  dim3 g3(512 / 16, 8 * 16);
  attn_fused<<<g3, 256, 0, stream>>>(Qh, Kh, Vh, mask, out);
}
